// MoETransformer_15298673508998
// MI455X (gfx1250) — matrix operand order repaired and checked by reading, not compiled, not run on hardware
//
#include <hip/hip_runtime.h>
#include <math.h>

// ---- problem constants (from reference) ----
#define PN 64
#define BN 32
#define NTOK (PN * BN)   // 2048 tokens
#define DD 512
#define HH 2048
#define EE 16
#define TM 32            // tokens per FFN tile (2 WMMA M-tiles)
#define MAX_TILE_BLOCKS 128   // 128 * 32 = 4096 = worst-case assignments per expert

typedef __bf16 bf16_t;
typedef __attribute__((ext_vector_type(16))) __bf16 v16bf;
typedef __attribute__((ext_vector_type(8)))  __bf16 v8bf;
typedef __attribute__((ext_vector_type(4)))  __bf16 v4bf;
typedef __attribute__((ext_vector_type(8)))  float  v8f;

union BF16x16 { v16bf v; v8bf h[2]; };

static __device__ inline v16bf make_frag(v8bf lo, v8bf hi) {
    BF16x16 u; u.h[0] = lo; u.h[1] = hi; return u.v;
}

// ---------------------------------------------------------------------------
// Kernel 0: zero output accumulator + routing counters (buffers are poisoned)
// ---------------------------------------------------------------------------
__global__ void moe_zero_kernel(float* __restrict__ out, int n_out,
                                int* __restrict__ cnt, float* __restrict__ imp) {
    int i = blockIdx.x * 256 + threadIdx.x;
    if (i < n_out) out[i] = 0.0f;
    if (i < EE) { cnt[i] = 0; imp[i] = 0.0f; }
}

// ---------------------------------------------------------------------------
// Kernel 1: noisy-top-k gating (eval mode): logits, top-2, softmax, routing
// ---------------------------------------------------------------------------
__global__ void moe_gate_kernel(const float* __restrict__ x,
                                const float* __restrict__ wg,
                                int* __restrict__ cnt, float* __restrict__ imp,
                                int* __restrict__ list, float* __restrict__ glist,
                                int maxt) {
    int t = blockIdx.x * blockDim.x + threadIdx.x;
    if (t >= NTOK) return;

    float v[EE];
#pragma unroll
    for (int e = 0; e < EE; ++e) v[e] = 0.0f;

    const float* xr = x + (size_t)t * DD;
    for (int d = 0; d < DD; ++d) {
        float xv = xr[d];
        const float* wr = wg + (size_t)d * EE;   // uniform address -> scalar loads
#pragma unroll
        for (int e = 0; e < EE; ++e) v[e] += xv * wr[e];
    }

    // top-2 (== first two of top-3; ties -> lowest index, matching lax.top_k)
    int i0 = 0; float m0 = v[0];
#pragma unroll
    for (int e = 1; e < EE; ++e) if (v[e] > m0) { m0 = v[e]; i0 = e; }
    int i1 = -1; float m1 = -1e30f;
#pragma unroll
    for (int e = 0; e < EE; ++e) if (e != i0 && v[e] > m1) { m1 = v[e]; i1 = e; }

    // softmax over the two kept logits
    float e1 = __expf(m1 - m0);
    float inv = 1.0f / (1.0f + e1);
    float g0 = inv, g1 = e1 * inv;

    int p0 = atomicAdd(&cnt[i0], 1);
    if (p0 < maxt) { list[i0 * maxt + p0] = t; glist[i0 * maxt + p0] = g0; }
    atomicAdd(&imp[i0], g0);

    int p1 = atomicAdd(&cnt[i1], 1);
    if (p1 < maxt) { list[i1 * maxt + p1] = t; glist[i1 * maxt + p1] = g1; }
    atomicAdd(&imp[i1], g1);
}

// ---------------------------------------------------------------------------
// stage a [32 rows x 512 cols] f32 weight slab into LDS, bf16, transposed so
// that B fragments are two contiguous 16B runs:  dst[col*32 + row].
// 4x4 blocking: 4 coalesced float4 row-loads -> 4 packed ds_store_b64
// (4 consecutive rows are contiguous in the transposed layout).
// Prefetches the slab two k-steps (+64 rows) ahead at 128B line granularity.
// ---------------------------------------------------------------------------
static __device__ inline void stage_slab(bf16_t* __restrict__ dst,
                                         const float* __restrict__ src,
                                         int stride, int tid) {
#pragma unroll
    for (int it = 0; it < 4; ++it) {
        int blk = it * 256 + tid;             // 1024 blocks: 8 rowgroups x 128 colgroups
        int r0 = (blk >> 7) << 2;
        int cc = (blk & 127) << 2;
        const float* p = src + (size_t)r0 * stride + cc;
        if ((tid & 7) == 0) {                  // one prefetch per 128B line per row
            __builtin_prefetch(p + (size_t)64 * stride, 0, 1);
            __builtin_prefetch(p + (size_t)65 * stride, 0, 1);
            __builtin_prefetch(p + (size_t)66 * stride, 0, 1);
            __builtin_prefetch(p + (size_t)67 * stride, 0, 1);
        }
        float4 f0 = *(const float4*)p;
        float4 f1 = *(const float4*)(p + (size_t)stride);
        float4 f2 = *(const float4*)(p + 2 * (size_t)stride);
        float4 f3 = *(const float4*)(p + 3 * (size_t)stride);
        v4bf q0 = { (bf16_t)f0.x, (bf16_t)f1.x, (bf16_t)f2.x, (bf16_t)f3.x };
        v4bf q1 = { (bf16_t)f0.y, (bf16_t)f1.y, (bf16_t)f2.y, (bf16_t)f3.y };
        v4bf q2 = { (bf16_t)f0.z, (bf16_t)f1.z, (bf16_t)f2.z, (bf16_t)f3.z };
        v4bf q3 = { (bf16_t)f0.w, (bf16_t)f1.w, (bf16_t)f2.w, (bf16_t)f3.w };
        *(v4bf*)(dst + (cc + 0) * 32 + r0) = q0;
        *(v4bf*)(dst + (cc + 1) * 32 + r0) = q1;
        *(v4bf*)(dst + (cc + 2) * 32 + r0) = q2;
        *(v4bf*)(dst + (cc + 3) * 32 + r0) = q3;
    }
}

// ---------------------------------------------------------------------------
// Kernel 2: grouped expert FFN.  One block = (expert e, tile of 32 tokens).
//   H = relu(X @ W1_e + b1_e)  ;  Y = H @ W2_e + b2_e  ;  out += gate * Y
// bf16 WMMA 16x16x32, double-buffered LDS weight slabs, reuse_a hints.
// ---------------------------------------------------------------------------
__global__ __launch_bounds__(256)
void moe_ffn_kernel(const float* __restrict__ x,
                    const float* __restrict__ w1, const float* __restrict__ b1,
                    const float* __restrict__ w2, const float* __restrict__ b2,
                    const int* __restrict__ cnt, const int* __restrict__ list,
                    const float* __restrict__ glist,
                    float* __restrict__ out, int maxt) {
    __shared__ __align__(16) bf16_t Xt[TM * DD];      //  32 KB gathered X tile
    __shared__ __align__(16) bf16_t Ht[TM * HH];      // 128 KB hidden tile
    __shared__ __align__(16) bf16_t Wt0[512 * 32];    //  32 KB W slab buf 0
    __shared__ __align__(16) bf16_t Wt1[512 * 32];    //  32 KB W slab buf 1
    __shared__ int   toks[TM];
    __shared__ float gts[TM];

    const int e    = blockIdx.y;
    const int tile = blockIdx.x;
    int c = cnt[e]; if (c > maxt) c = maxt;
    if (tile * TM >= c) return;                       // uniform early-exit

    const int tid  = threadIdx.x;
    const int wave = tid >> 5;
    const int lane = tid & 31;
    const int half = lane >> 4;                       // lane group (0:0-15, 1:16-31)
    const int lm   = lane & 15;

    if (tid < TM) {
        int idx = tile * TM + tid;
        if (idx < c) { toks[tid] = list[e * maxt + idx]; gts[tid] = glist[e * maxt + idx]; }
        else         { toks[tid] = 0;                    gts[tid] = 0.0f; }  // padded row
    }
    __syncthreads();

    // gather 32 token rows of X, convert f32 -> bf16 (8 threads per row)
    {
        int row = tid >> 3, seg = tid & 7;
        const float* src = x + (size_t)toks[row] * DD + seg * 64;
        bf16_t* dst = Xt + row * DD + seg * 64;
#pragma unroll
        for (int i = 0; i < 64; i += 4) {
            float4 f = *(const float4*)(src + i);
            dst[i + 0] = (bf16_t)f.x; dst[i + 1] = (bf16_t)f.y;
            dst[i + 2] = (bf16_t)f.z; dst[i + 3] = (bf16_t)f.w;
        }
    }

    const float* W1e = w1 + (size_t)e * DD * HH;
    const float* W2e = w2 + (size_t)e * HH * DD;

    // ------------- stage 1: H = relu(X @ W1 + b1), in four 512-col quarters
    for (int q = 0; q < 4; ++q) {
        const int nbase = q * 512;
        v8f acc[2][4];
#pragma unroll
        for (int mt = 0; mt < 2; ++mt)
#pragma unroll
            for (int nt = 0; nt < 4; ++nt)
#pragma unroll
                for (int r = 0; r < 8; ++r) acc[mt][nt][r] = 0.0f;

        __syncthreads();
        stage_slab(Wt0, W1e + nbase, HH, tid);        // prologue: k-step 0

        for (int ks = 0; ks < DD / 32; ++ks) {
            const int k0 = ks * 32;
            __syncthreads();   // slab(ks) staged; slab(ks+1) buffer free
            if (ks < DD / 32 - 1)
                stage_slab((ks & 1) ? Wt0 : Wt1,
                           W1e + (size_t)(ks + 1) * 32 * HH + nbase, HH, tid);
            const bf16_t* slab = (ks & 1) ? Wt1 : Wt0;

            // preload all B fragments, then A fragments, then WMMA chain
            v16bf bfr[4];
#pragma unroll
            for (int nt = 0; nt < 4; ++nt) {
                const bf16_t* br = slab + ((wave * 4 + nt) * 16 + lm) * 32 + 8 * half;
                bfr[nt] = make_frag(*(const v8bf*)br, *(const v8bf*)(br + 16));
            }
            const bf16_t* ar0 = Xt + lm * DD + k0 + 8 * half;
            const bf16_t* ar1 = ar0 + 16 * DD;
            v16bf a0 = make_frag(*(const v8bf*)ar0, *(const v8bf*)(ar0 + 16));
            v16bf a1 = make_frag(*(const v8bf*)ar1, *(const v8bf*)(ar1 + 16));

            acc[0][0] = __builtin_amdgcn_wmma_f32_16x16x32_bf16(
                false, a0, false, bfr[0], (short)0, acc[0][0], false, false);
#pragma unroll
            for (int nt = 1; nt < 4; ++nt)
                acc[0][nt] = __builtin_amdgcn_wmma_f32_16x16x32_bf16(
                    false, a0, false, bfr[nt], (short)0, acc[0][nt], true, false);
            acc[1][0] = __builtin_amdgcn_wmma_f32_16x16x32_bf16(
                false, a1, false, bfr[0], (short)0, acc[1][0], false, false);
#pragma unroll
            for (int nt = 1; nt < 4; ++nt)
                acc[1][nt] = __builtin_amdgcn_wmma_f32_16x16x32_bf16(
                    false, a1, false, bfr[nt], (short)0, acc[1][nt], true, false);
        }

        // epilogue: bias + relu, bf16 into Ht  (C layout: m = r + 8*half, n = lm)
#pragma unroll
        for (int nt = 0; nt < 4; ++nt) {
            int ncol = nbase + (wave * 4 + nt) * 16 + lm;
            float bb = b1[e * HH + ncol];
#pragma unroll
            for (int mt = 0; mt < 2; ++mt) {
                v8f v = acc[mt][nt];
#pragma unroll
                for (int r = 0; r < 8; ++r) {
                    int m = mt * 16 + r + 8 * half;
                    float hv = v[r] + bb;
                    Ht[m * HH + ncol] = (bf16_t)(hv > 0.0f ? hv : 0.0f);
                }
            }
        }
    }

    // ------------- stage 2: Y = H @ W2 + b2, gate-scaled scatter-add
    v8f acc2[2][4];
#pragma unroll
    for (int mt = 0; mt < 2; ++mt)
#pragma unroll
        for (int nt = 0; nt < 4; ++nt)
#pragma unroll
            for (int r = 0; r < 8; ++r) acc2[mt][nt][r] = 0.0f;

    __syncthreads();                                  // Ht complete, Wt0 free
    stage_slab(Wt0, W2e, DD, tid);                    // prologue: k-step 0

    for (int ks = 0; ks < HH / 32; ++ks) {
        const int k0 = ks * 32;
        __syncthreads();
        if (ks < HH / 32 - 1)
            stage_slab((ks & 1) ? Wt0 : Wt1,
                       W2e + (size_t)(ks + 1) * 32 * DD, DD, tid);
        const bf16_t* slab = (ks & 1) ? Wt1 : Wt0;

        v16bf bfr[4];
#pragma unroll
        for (int nt = 0; nt < 4; ++nt) {
            const bf16_t* br = slab + ((wave * 4 + nt) * 16 + lm) * 32 + 8 * half;
            bfr[nt] = make_frag(*(const v8bf*)br, *(const v8bf*)(br + 16));
        }
        const bf16_t* ar0 = Ht + lm * HH + k0 + 8 * half;
        const bf16_t* ar1 = ar0 + 16 * HH;
        v16bf a0 = make_frag(*(const v8bf*)ar0, *(const v8bf*)(ar0 + 16));
        v16bf a1 = make_frag(*(const v8bf*)ar1, *(const v8bf*)(ar1 + 16));

        acc2[0][0] = __builtin_amdgcn_wmma_f32_16x16x32_bf16(
            false, a0, false, bfr[0], (short)0, acc2[0][0], false, false);
#pragma unroll
        for (int nt = 1; nt < 4; ++nt)
            acc2[0][nt] = __builtin_amdgcn_wmma_f32_16x16x32_bf16(
                false, a0, false, bfr[nt], (short)0, acc2[0][nt], true, false);
        acc2[1][0] = __builtin_amdgcn_wmma_f32_16x16x32_bf16(
            false, a1, false, bfr[0], (short)0, acc2[1][0], false, false);
#pragma unroll
        for (int nt = 1; nt < 4; ++nt)
            acc2[1][nt] = __builtin_amdgcn_wmma_f32_16x16x32_bf16(
                false, a1, false, bfr[nt], (short)0, acc2[1][nt], true, false);
    }

#pragma unroll
    for (int nt = 0; nt < 4; ++nt) {
        int ncol = (wave * 4 + nt) * 16 + lm;
        float bb = b2[e * DD + ncol];
#pragma unroll
        for (int mt = 0; mt < 2; ++mt) {
            v8f v = acc2[mt][nt];
#pragma unroll
            for (int r = 0; r < 8; ++r) {
                int m = mt * 16 + r + 8 * half;
                float val = gts[m] * (v[r] + bb);     // gate==0 kills padded rows
                atomicAdd(&out[(size_t)toks[m] * DD + ncol], val);
            }
        }
    }
}

// ---------------------------------------------------------------------------
// Kernel 3: aux load-balancing loss (unbiased variance of importance)
// ---------------------------------------------------------------------------
__global__ void moe_aux_kernel(const float* __restrict__ imp, float* __restrict__ out_aux) {
    if (threadIdx.x == 0) {
        float s = 0.0f;
#pragma unroll
        for (int e = 0; e < EE; ++e) s += imp[e];
        float mean = s * (1.0f / EE);
        float var = 0.0f;
#pragma unroll
        for (int e = 0; e < EE; ++e) { float d = imp[e] - mean; var += d * d; }
        var *= (1.0f / (EE - 1));
        out_aux[0] = 0.01f * var / (mean * mean + 1e-10f);
    }
}

// ---------------------------------------------------------------------------
extern "C" void kernel_launch(void* const* d_in, const int* in_sizes, int n_in,
                              void* d_out, int out_size, void* d_ws, size_t ws_size,
                              hipStream_t stream) {
    const float* x  = (const float*)d_in[0];
    const float* wg = (const float*)d_in[1];
    const float* w1 = (const float*)d_in[2];
    const float* b1 = (const float*)d_in[3];
    const float* w2 = (const float*)d_in[4];
    const float* b2 = (const float*)d_in[5];
    float* out = (float*)d_out;             // [NTOK*DD] out, then [1] aux_loss

    // workspace layout: cnt[16] | imp[16] | list[E*maxt] | glist[E*maxt]
    char* ws = (char*)d_ws;
    int*   cnt = (int*)ws;
    float* imp = (float*)(ws + 64);
    size_t avail = (ws_size > 128) ? (ws_size - 128) : 0;
    int maxt = (int)(avail / (size_t)(EE * 8));      // 4B list + 4B gate per slot
    if (maxt > NTOK * 2) maxt = NTOK * 2;            // 4096 = absolute worst case
    if (maxt < 1) maxt = 1;
    int*   list  = (int*)(ws + 128);
    float* glist = (float*)(ws + 128 + (size_t)EE * maxt * 4);

    moe_zero_kernel<<<(NTOK * DD + 255) / 256, 256, 0, stream>>>(out, NTOK * DD, cnt, imp);
    moe_gate_kernel<<<NTOK / 256, 256, 0, stream>>>(x, wg, cnt, imp, list, glist, maxt);
    dim3 grid(MAX_TILE_BLOCKS, EE);
    moe_ffn_kernel<<<grid, 256, 0, stream>>>(x, w1, b1, w2, b2, cnt, list, glist, out, maxt);
    moe_aux_kernel<<<1, 32, 0, stream>>>(imp, out + (size_t)NTOK * DD);
}
